// CustomMultiHeadAttentionStoichRoPE_36249523978453
// MI455X (gfx1250) — matrix operand order repaired and checked by reading, not compiled, not run on hardware
//
#include <hip/hip_runtime.h>
#include <math.h>

typedef __attribute__((ext_vector_type(16))) _Float16 v16h;
typedef __attribute__((ext_vector_type(8)))  _Float16 v8h;
typedef __attribute__((ext_vector_type(8)))  float    v8f;

#define D_MODEL  1024
#define NHEAD    16
#define HEAD_DIM 64
#define HALF_DIM 32
#define B_SZ     4
#define T_SZ     2048
#define M_ROWS   (B_SZ * T_SZ)   // 8192

static __device__ __forceinline__ v16h make_v16(v8h lo, v8h hi) {
  v16h r;
#pragma unroll
  for (int i = 0; i < 8; ++i) { r[i] = lo[i]; r[i + 8] = hi[i]; }
  return r;
}

static __device__ __forceinline__ v8f wmma16(v16h a, v16h b, v8f c) {
  // D = A(16x32 f16) * B(32x16 f16) + C(16x16 f32)
  return __builtin_amdgcn_wmma_f32_16x16x32_f16(false, a, false, b,
                                                (short)0, c, false, false);
}

// ---------------------------------------------------------------- convert
__global__ void cvt_f32_to_f16(const float* __restrict__ src,
                               _Float16* __restrict__ dst, int n) {
  int i = blockIdx.x * blockDim.x + threadIdx.x;
  if (i < n) dst[i] = (_Float16)src[i];
}

// ---------------------------------------------------------------- GEMM
// out = A(MxK f16) @ W^T + bias, W stored row-major NxK (so B-fragment
// columns of W^T are contiguous rows of W). Wave tile 64x64 (4x4 WMMA
// fragments): 16 v_wmma per 16 b128 fragment loads per K-step.
// MODE 0: f32 row-major [M][N]           -> outF
// MODE 1: f16 scatter   [B,H,HEAD_DIM,T] -> outH  (V transposed)
// MODE 2: f32 scatter   [B,H,T,HEAD_DIM] -> outF  (Q/K pre-RoPE)
template <int MODE>
__global__ __launch_bounds__(128)
void gemm_xwT_wmma(const _Float16* __restrict__ A16,
                   const _Float16* __restrict__ W16,
                   const float* __restrict__ bias,
                   float* __restrict__ outF,
                   _Float16* __restrict__ outH) {
  const int K = D_MODEL, N = D_MODEL;
  const int lane = threadIdx.x & 31;
  const int wave = threadIdx.x >> 5;
  const int l15  = lane & 15;
  const int hi   = lane >> 4;
  // 4 waves arranged 2x2 -> block tile 128(M) x 128(N)
  const int m0 = blockIdx.x * 128 + (wave >> 1) * 64;
  const int n0 = blockIdx.y * 128 + (wave & 1) * 64;

  v8f acc[4][4];
  v8f zero = {};
#pragma unroll
  for (int i = 0; i < 4; ++i)
#pragma unroll
    for (int j = 0; j < 4; ++j) acc[i][j] = zero;

  const _Float16* aBase = A16 + (size_t)(m0 + l15) * K + 8 * hi;
  const _Float16* bBase = W16 + (size_t)(n0 + l15) * K + 8 * hi;

  for (int k = 0; k < K; k += 32) {
    v16h a[4], b[4];
#pragma unroll
    for (int mt = 0; mt < 4; ++mt) {
      const _Float16* ap = aBase + (size_t)(mt * 16) * K + k;
      a[mt] = make_v16(*(const v8h*)ap, *(const v8h*)(ap + 16));
    }
#pragma unroll
    for (int nt = 0; nt < 4; ++nt) {
      const _Float16* bp = bBase + (size_t)(nt * 16) * K + k;
      b[nt] = make_v16(*(const v8h*)bp, *(const v8h*)(bp + 16));
    }
    __builtin_prefetch(aBase + k + 32, 0, 3);  // streaming prefetch
    __builtin_prefetch(bBase + k + 32, 0, 3);
#pragma unroll
    for (int mt = 0; mt < 4; ++mt)
#pragma unroll
      for (int nt = 0; nt < 4; ++nt)
        acc[mt][nt] = wmma16(a[mt], b[nt], acc[mt][nt]);
  }

#pragma unroll
  for (int nt = 0; nt < 4; ++nt) {
    const int n = n0 + nt * 16 + l15;
    const float bs = bias[n];
    const int h = n >> 6, d = n & 63;
#pragma unroll
    for (int mt = 0; mt < 4; ++mt) {
#pragma unroll
      for (int r = 0; r < 8; ++r) {
        int m = m0 + mt * 16 + r + 8 * hi;   // C-layout: m = r + 8*half
        float v = acc[mt][nt][r] + bs;
        if (MODE == 0) {
          outF[(size_t)m * N + n] = v;
        } else {
          int bb = m >> 11, t = m & (T_SZ - 1);
          if (MODE == 2)
            outF[(((size_t)bb * NHEAD + h) * T_SZ + t) * HEAD_DIM + d] = v;
          else
            outH[(((size_t)bb * NHEAD + h) * HEAD_DIM + d) * T_SZ + t] = (_Float16)v;
        }
      }
    }
  }
}

// ---------------------------------------------------------------- RoPE
__global__ void rope_f32_to_f16(const float* __restrict__ Qf,
                                const float* __restrict__ Kf,
                                const float* __restrict__ frac,
                                _Float16* __restrict__ Qh,
                                _Float16* __restrict__ Kh) {
  int idx = blockIdx.x * blockDim.x + threadIdx.x;
  if (idx >= B_SZ * NHEAD * T_SZ * HALF_DIM) return;
  int i = idx & 31;
  int t = (idx >> 5) & (T_SZ - 1);
  int h = (idx >> 16) & (NHEAD - 1);
  int b = idx >> 20;
  float pos = frac[(size_t)b * T_SZ + t] * 1000.0f;
  float div = powf(10000.0f, (float)i * (1.0f / 32.0f));
  float ang = pos / div;
  float c = cosf(ang), s = sinf(ang);
  size_t base = (((size_t)b * NHEAD + h) * T_SZ + t) * HEAD_DIM;
  float ql = Qf[base + i], qr = Qf[base + 32 + i];
  float kl = Kf[base + i], kr = Kf[base + 32 + i];
  Qh[base + i]      = (_Float16)(ql * c - qr * s);
  Qh[base + 32 + i] = (_Float16)(ql * s + qr * c);
  Kh[base + i]      = (_Float16)(kl * c - kr * s);
  Kh[base + 32 + i] = (_Float16)(kl * s + kr * c);
}

// ---------------------------------------------------------------- flash attn
// One block = 4 waves = 64 queries of one (b,h); each wave owns 16 queries.
// Streams K/V in 32-wide tiles with online softmax; all matmuls via WMMA.
__global__ __launch_bounds__(128)
void flash_attn_wmma(const _Float16* __restrict__ Qh,
                     const _Float16* __restrict__ Kh,
                     const _Float16* __restrict__ Vt,
                     _Float16* __restrict__ AO) {
  __shared__ _Float16 Plds[4][16][40];   // per-wave 16x32 P tile, padded
  const int lane = threadIdx.x & 31;
  const int wave = threadIdx.x >> 5;
  const int l15  = lane & 15;
  const int hi   = lane >> 4;
  const int bh = blockIdx.x;
  const int b  = bh >> 4;
  const int h  = bh & (NHEAD - 1);
  const int q0 = blockIdx.y * 64 + wave * 16;

  const _Float16* Qbase = Qh + (size_t)bh * T_SZ * HEAD_DIM;
  const _Float16* Kbase = Kh + (size_t)bh * T_SZ * HEAD_DIM;
  const _Float16* Vbase = Vt + (size_t)bh * HEAD_DIM * T_SZ;

  // Persistent Q fragments: d=0..31 and d=32..63
  const _Float16* qp = Qbase + (size_t)(q0 + l15) * HEAD_DIM;
  v16h aQ0 = make_v16(*(const v8h*)(qp + 8 * hi),
                      *(const v8h*)(qp + 16 + 8 * hi));
  v16h aQ1 = make_v16(*(const v8h*)(qp + 32 + 8 * hi),
                      *(const v8h*)(qp + 48 + 8 * hi));

  v8f o0 = {}, o1 = {}, o2 = {}, o3 = {};
  float mrow[8], lrow[8];
#pragma unroll
  for (int r = 0; r < 8; ++r) { mrow[r] = -__builtin_inff(); lrow[r] = 0.f; }

  for (int j = 0; j < T_SZ; j += 32) {
    // K B-fragments: col = key index (j..j+31), K-dim = head dim
    const _Float16* kp0 = Kbase + (size_t)(j + l15) * HEAD_DIM + 8 * hi;
    const _Float16* kp1 = kp0 + 16 * HEAD_DIM;
    v16h bK00 = make_v16(*(const v8h*)(kp0), *(const v8h*)(kp0 + 16));        // keys j..j+15,  d 0..31
    v16h bK01 = make_v16(*(const v8h*)(kp0 + 32), *(const v8h*)(kp0 + 48));   // keys j..j+15,  d 32..63
    v16h bK10 = make_v16(*(const v8h*)(kp1), *(const v8h*)(kp1 + 16));        // keys j+16..31, d 0..31
    v16h bK11 = make_v16(*(const v8h*)(kp1 + 32), *(const v8h*)(kp1 + 48));

    v8f s0 = {}, s1 = {};
    s0 = wmma16(aQ0, bK00, s0); s0 = wmma16(aQ1, bK01, s0);
    s1 = wmma16(aQ0, bK10, s1); s1 = wmma16(aQ1, bK11, s1);
    s0 *= 0.125f;                 // d^{-1/2}
    s1 *= 0.125f;

#pragma unroll
    for (int r = 0; r < 8; ++r) {
      float mx = fmaxf(s0[r], s1[r]);
#pragma unroll
      for (int off = 1; off < 16; off <<= 1)
        mx = fmaxf(mx, __shfl_xor(mx, off, 32));   // stays in 16-lane half
      float mn = fmaxf(mrow[r], mx);
      float sc = __expf(mrow[r] - mn);
      float p0 = __expf(s0[r] - mn);
      float p1 = __expf(s1[r] - mn);
      float ps = p0 + p1;
#pragma unroll
      for (int off = 1; off < 16; off <<= 1)
        ps += __shfl_xor(ps, off, 32);
      lrow[r] = lrow[r] * sc + ps;
      mrow[r] = mn;
      int row = r + 8 * hi;                  // C-layout row
      Plds[wave][row][l15]      = (_Float16)p0;
      Plds[wave][row][16 + l15] = (_Float16)p1;
      o0[r] *= sc; o1[r] *= sc; o2[r] *= sc; o3[r] *= sc;
    }

    // Re-read P as an A-fragment (same-wave LDS ops are in-order)
    const _Float16* pp = &Plds[wave][l15][0];
    v16h aP = make_v16(*(const v8h*)(pp + 8 * hi),
                       *(const v8h*)(pp + 16 + 8 * hi));

    // V^T B-fragments: col = head-dim, K-dim = key index (contiguous in Vt)
    const _Float16* vp = Vbase + (size_t)l15 * T_SZ + j + 8 * hi;
    v16h bV0 = make_v16(*(const v8h*)(vp), *(const v8h*)(vp + 16));
    v16h bV1 = make_v16(*(const v8h*)(vp + (size_t)16 * T_SZ), *(const v8h*)(vp + (size_t)16 * T_SZ + 16));
    v16h bV2 = make_v16(*(const v8h*)(vp + (size_t)32 * T_SZ), *(const v8h*)(vp + (size_t)32 * T_SZ + 16));
    v16h bV3 = make_v16(*(const v8h*)(vp + (size_t)48 * T_SZ), *(const v8h*)(vp + (size_t)48 * T_SZ + 16));
    o0 = wmma16(aP, bV0, o0);
    o1 = wmma16(aP, bV1, o1);
    o2 = wmma16(aP, bV2, o2);
    o3 = wmma16(aP, bV3, o3);
  }

  // Normalize and scatter to [B,T,H*d] f16 for the output projection
  _Float16* aob = AO + (size_t)b * T_SZ * D_MODEL + (size_t)h * HEAD_DIM;
#pragma unroll
  for (int r = 0; r < 8; ++r) {
    float inv = 1.0f / lrow[r];
    int t = q0 + r + 8 * hi;
    _Float16* op = aob + (size_t)t * D_MODEL + l15;
    op[0]  = (_Float16)(o0[r] * inv);
    op[16] = (_Float16)(o1[r] * inv);
    op[32] = (_Float16)(o2[r] * inv);
    op[48] = (_Float16)(o3[r] * inv);
  }
}

// ---------------------------------------------------------------- launch
extern "C" void kernel_launch(void* const* d_in, const int* in_sizes, int n_in,
                              void* d_out, int out_size, void* d_ws, size_t ws_size,
                              hipStream_t stream) {
  (void)in_sizes; (void)n_in; (void)out_size; (void)ws_size;
  const float* x    = (const float*)d_in[0];
  const float* frac = (const float*)d_in[1];
  const float* Wq   = (const float*)d_in[2];
  const float* bq   = (const float*)d_in[3];
  const float* Wk   = (const float*)d_in[4];
  const float* bk   = (const float*)d_in[5];
  const float* Wv   = (const float*)d_in[6];
  const float* bv   = (const float*)d_in[7];
  const float* Wo   = (const float*)d_in[8];
  const float* bo   = (const float*)d_in[9];
  float* out = (float*)d_out;

  char* ws = (char*)d_ws;
  size_t off = 0;
  auto take = [&](size_t bytes) -> char* {
    char* p = ws + off; off += (bytes + 255) & ~(size_t)255; return p;
  };
  _Float16* x16  = (_Float16*)take((size_t)M_ROWS * D_MODEL * 2);
  _Float16* Wq16 = (_Float16*)take((size_t)D_MODEL * D_MODEL * 2);
  _Float16* Wk16 = (_Float16*)take((size_t)D_MODEL * D_MODEL * 2);
  _Float16* Wv16 = (_Float16*)take((size_t)D_MODEL * D_MODEL * 2);
  _Float16* Wo16 = (_Float16*)take((size_t)D_MODEL * D_MODEL * 2);
  float*    Qf   = (float*)take((size_t)M_ROWS * D_MODEL * 4);
  float*    Kf   = (float*)take((size_t)M_ROWS * D_MODEL * 4);
  _Float16* QhB  = (_Float16*)take((size_t)M_ROWS * D_MODEL * 2);
  _Float16* KhB  = (_Float16*)take((size_t)M_ROWS * D_MODEL * 2);
  _Float16* VtB  = (_Float16*)take((size_t)M_ROWS * D_MODEL * 2);
  _Float16* AO16 = (_Float16*)Qf;   // reuse: Qf dead after RoPE

  const int nX = M_ROWS * D_MODEL;
  const int nW = D_MODEL * D_MODEL;
  cvt_f32_to_f16<<<(nX + 255) / 256, 256, 0, stream>>>(x,  x16,  nX);
  cvt_f32_to_f16<<<(nW + 255) / 256, 256, 0, stream>>>(Wq, Wq16, nW);
  cvt_f32_to_f16<<<(nW + 255) / 256, 256, 0, stream>>>(Wk, Wk16, nW);
  cvt_f32_to_f16<<<(nW + 255) / 256, 256, 0, stream>>>(Wv, Wv16, nW);
  cvt_f32_to_f16<<<(nW + 255) / 256, 256, 0, stream>>>(Wo, Wo16, nW);

  dim3 ggrid(M_ROWS / 128, D_MODEL / 128);   // 64 x 8 blocks, 128 thr each
  gemm_xwT_wmma<2><<<ggrid, 128, 0, stream>>>(x16, Wq16, bq, Qf, nullptr);
  gemm_xwT_wmma<2><<<ggrid, 128, 0, stream>>>(x16, Wk16, bk, Kf, nullptr);
  gemm_xwT_wmma<1><<<ggrid, 128, 0, stream>>>(x16, Wv16, bv, nullptr, VtB);

  const int nR = B_SZ * NHEAD * T_SZ * HALF_DIM;
  rope_f32_to_f16<<<(nR + 255) / 256, 256, 0, stream>>>(Qf, Kf, frac, QhB, KhB);

  flash_attn_wmma<<<dim3(B_SZ * NHEAD, T_SZ / 64), 128, 0, stream>>>(QhB, KhB, VtB, AO16);

  gemm_xwT_wmma<0><<<ggrid, 128, 0, stream>>>(AO16, Wo16, bo, out, nullptr);
}
